// AttentionBlock_6932077216169
// MI455X (gfx1250) — compile-verified, matrix operand order checked by reading, not run on hardware
//
#include <hip/hip_runtime.h>
#include <hip/hip_bf16.h>

// CDNA5 / gfx1250 attention block: GroupNorm -> QKV GEMM -> flash attention
// -> proj GEMM + residual.  All matmuls via V_WMMA_F32_16X16X32_BF16.
// bf16 tile staging uses GLOBAL_LOAD_ASYNC_TO_LDS_B128 when the builtin exists.

typedef __bf16  bf16_t;
typedef __bf16  v16bf __attribute__((ext_vector_type(16)));
typedef float   v8f   __attribute__((ext_vector_type(8)));

#define C_DIM   512
#define L_DIM   4096
#define HD_DIM  128
#define NH_DIM  4
#define B_DIM   2

#if defined(__AMDGCN__) && \
    __has_builtin(__builtin_amdgcn_global_load_async_to_lds_b128) && \
    __has_builtin(__builtin_amdgcn_s_wait_asynccnt)
#define HAVE_ASYNC_LDS 1
typedef int v4i __attribute__((ext_vector_type(4)));
typedef __attribute__((address_space(1))) v4i as1_v4i;   // global int4*
typedef __attribute__((address_space(3))) v4i as3_v4i;   // LDS int4*
#else
#define HAVE_ASYNC_LDS 0
#endif

__device__ __forceinline__ v8f v8f_zero() {
    v8f z = {0.f,0.f,0.f,0.f,0.f,0.f,0.f,0.f};
    return z;
}

__device__ __forceinline__ v8f wmma_bf16(v16bf a, v16bf b, v8f c) {
    return __builtin_amdgcn_wmma_f32_16x16x32_bf16(false, a, false, b,
                                                   (short)0, c, false, false);
}

// ---------------------------------------------------------------------------
// Kernel 1: GroupNorm.  One block per (group, batch).  Group = 16 ch x 4096.
// Writes normalized activations TRANSPOSED as bf16 [b, L, C]; for each l the
// group's 16 channels are contiguous, so stores are 2 x 16B per row element.
// ---------------------------------------------------------------------------
__global__ void __launch_bounds__(256)
groupnorm_kernel(const float* __restrict__ x,
                 const float* __restrict__ gn_w,
                 const float* __restrict__ gn_b,
                 bf16_t* __restrict__ xn)
{
    const int g  = blockIdx.x;          // 0..31
    const int b  = blockIdx.y;          // 0..1
    const int tid = threadIdx.x;
    const int GELEMS = 16 * L_DIM;      // 65536

    const float* xg = x + ((size_t)b * C_DIM + g * 16) * L_DIM;

    float s = 0.f, s2 = 0.f;
    for (int i = tid; i < GELEMS; i += 256) {
        float v = xg[i];
        s += v; s2 += v * v;
    }
    __shared__ float rs[256], rs2[256];
    rs[tid] = s; rs2[tid] = s2;
    __syncthreads();
    for (int off = 128; off > 0; off >>= 1) {
        if (tid < off) { rs[tid] += rs[tid + off]; rs2[tid] += rs2[tid + off]; }
        __syncthreads();
    }
    const float inv_n = 1.0f / (float)GELEMS;
    const float mu  = rs[0] * inv_n;
    const float var = rs2[0] * inv_n - mu * mu;
    const float rstd = rsqrtf(var + 1e-5f);

    float wg[16], bg[16];
#pragma unroll
    for (int c = 0; c < 16; ++c) {
        wg[c] = gn_w[g * 16 + c];
        bg[c] = gn_b[g * 16 + c];
    }

    for (int l = tid; l < L_DIM; l += 256) {
        union { bf16_t h[16]; uint4 u[2]; } o;
#pragma unroll
        for (int c = 0; c < 16; ++c)
            o.h[c] = (bf16_t)((xg[(size_t)c * L_DIM + l] - mu) * rstd * wg[c] + bg[c]);
        uint4* dst = (uint4*)(xn + ((size_t)b * L_DIM + l) * C_DIM + g * 16);
        dst[0] = o.u[0];
        dst[1] = o.u[1];
    }
}

// ---------------------------------------------------------------------------
// Vectorized 128x32 tile staging into LDS (bf16), 256 threads.
// ---------------------------------------------------------------------------
__device__ __forceinline__ void fill_tile(bf16_t* __restrict__ dst,
                                          const float* __restrict__ src,
                                          int K, int row0, int k0, int tid)
{
    float4 v[4];
#pragma unroll
    for (int i = 0; i < 4; ++i) {                  // 1024 chunks of 4 floats
        int q = tid + i * 256;
        int row = q >> 3, kp = (q & 7) << 2;
        v[i] = *(const float4*)(src + (size_t)(row0 + row) * K + k0 + kp);
    }
#pragma unroll
    for (int i = 0; i < 4; ++i) {
        int q = tid + i * 256;
        int row = q >> 3, kp = (q & 7) << 2;
        union { bf16_t h[4]; uint2 u; } t;
        t.h[0] = (bf16_t)v[i].x; t.h[1] = (bf16_t)v[i].y;
        t.h[2] = (bf16_t)v[i].z; t.h[3] = (bf16_t)v[i].w;
        *(uint2*)(dst + row * 32 + kp) = t.u;
    }
}

__device__ __forceinline__ void fill_tile(bf16_t* __restrict__ dst,
                                          const bf16_t* __restrict__ src,
                                          int K, int row0, int k0, int tid)
{
#if HAVE_ASYNC_LDS
#pragma unroll
    for (int i = 0; i < 2; ++i) {                  // 512 chunks of 8 bf16
        int q = tid + i * 256;
        int row = q >> 2, kp = (q & 3) << 3;
        __builtin_amdgcn_global_load_async_to_lds_b128(
            (as1_v4i*)(src + (size_t)(row0 + row) * K + k0 + kp),
            (as3_v4i*)(dst + row * 32 + kp), 0, 0);
    }
    __builtin_amdgcn_s_wait_asynccnt(0);
#else
    uint4 v[2];
#pragma unroll
    for (int i = 0; i < 2; ++i) {
        int q = tid + i * 256;
        int row = q >> 2, kp = (q & 3) << 3;
        v[i] = *(const uint4*)(src + (size_t)(row0 + row) * K + k0 + kp);
    }
#pragma unroll
    for (int i = 0; i < 2; ++i) {
        int q = tid + i * 256;
        int row = q >> 2, kp = (q & 3) << 3;
        *(uint4*)(dst + row * 32 + kp) = v[i];
    }
#endif
}

// ---------------------------------------------------------------------------
// Shared WMMA GEMM core:  D[m0..+128, n0..+128] = A[M,K] * B[N,K]^T
// Block = 256 threads = 8 waves in a 4(M) x 2(N) grid; each wave owns a 32x64
// sub-tile = 2x4 WMMA accumulators.  BK = 32 (one bf16 WMMA per K step).
// ---------------------------------------------------------------------------
template <typename TB>
__device__ __forceinline__ void gemm_tile(const float* __restrict__ A,
                                          const TB* __restrict__ B,
                                          int K, int m0, int n0,
                                          v8f acc[2][4],
                                          bf16_t* As, bf16_t* Bs)
{
    const int tid  = threadIdx.x;
    const int lane = tid & 31;
    const int wave = tid >> 5;
    const int wm = wave >> 1, wn = wave & 1;
    const int ln = lane & 15, hi = lane >> 4;

#pragma unroll
    for (int i = 0; i < 2; ++i)
#pragma unroll
        for (int j = 0; j < 4; ++j) acc[i][j] = v8f_zero();

    for (int k0 = 0; k0 < K; k0 += 32) {
        __syncthreads();
        fill_tile(As, A, K, m0, k0, tid);
        fill_tile(Bs, B, K, n0, k0, tid);
        __syncthreads();
#pragma unroll
        for (int sm = 0; sm < 2; ++sm) {
            // A frag: lanes 0-15 row m, K 0-15; lanes 16-31 same rows, K 16-31
            v16bf a = *(const v16bf*)(As + (wm * 32 + sm * 16 + ln) * 32 + hi * 16);
#pragma unroll
            for (int sn = 0; sn < 4; ++sn) {
                v16bf b = *(const v16bf*)(Bs + (wn * 64 + sn * 16 + ln) * 32 + hi * 16);
                acc[sm][sn] = wmma_bf16(a, b, acc[sm][sn]);
            }
        }
    }
    __syncthreads();
}

// ---------------------------------------------------------------------------
// Kernel 2: QKV GEMM.  qkv[o,l] = sum_c w_qkv[o,c]*xn[l,c] + b_qkv[o]
// Epilogue scatters: Q,K -> [b,nh,L,hd] bf16 (Q pre-scaled by 1/sqrt(hd)),
//                    V   -> [b,nh,hd,L] bf16.
// ---------------------------------------------------------------------------
__global__ void __launch_bounds__(256)
qkv_gemm_kernel(const float* __restrict__ w_qkv,
                const bf16_t* __restrict__ xn,
                const float* __restrict__ b_qkv,
                bf16_t* __restrict__ qb,
                bf16_t* __restrict__ kb,
                bf16_t* __restrict__ vb)
{
    __shared__ __align__(32) bf16_t As[128 * 32];
    __shared__ __align__(32) bf16_t Bs[128 * 32];
    const int n0 = blockIdx.x * 128;
    const int m0 = blockIdx.y * 128;
    const int batch = blockIdx.z;

    v8f acc[2][4];
    gemm_tile(w_qkv, xn + (size_t)batch * L_DIM * C_DIM, C_DIM, m0, n0, acc, As, Bs);

    const int lane = threadIdx.x & 31;
    const int wave = threadIdx.x >> 5;
    const int wm = wave >> 1, wn = wave & 1;
    const int ln = lane & 15, hi = lane >> 4;
    const float qscale = 0.08838834764831845f;   // 1/sqrt(128)

#pragma unroll
    for (int sm = 0; sm < 2; ++sm)
#pragma unroll
        for (int sn = 0; sn < 4; ++sn)
#pragma unroll
            for (int r = 0; r < 8; ++r) {
                int o = m0 + wm * 32 + sm * 16 + r + 8 * hi;
                int l = n0 + wn * 64 + sn * 16 + ln;
                float val = acc[sm][sn][r] + b_qkv[o];
                int sect = o >> 9;               // 0=q 1=k 2=v
                int oc = o & (C_DIM - 1);
                int h  = oc >> 7;
                int c  = oc & (HD_DIM - 1);
                size_t bh = (size_t)batch * NH_DIM + h;
                if (sect == 0)
                    qb[(bh * L_DIM + l) * HD_DIM + c] = (bf16_t)(val * qscale);
                else if (sect == 1)
                    kb[(bh * L_DIM + l) * HD_DIM + c] = (bf16_t)val;
                else
                    vb[(bh * HD_DIM + c) * L_DIM + l] = (bf16_t)val;
            }
}

// ---------------------------------------------------------------------------
// Kernel 3: flash attention.  Block = (128 query rows, head, batch); 8 waves,
// each owns 16 query rows.  Key loop in tiles of 64; K/V staged in LDS via
// async global->LDS copies, Q fragments held in registers for the whole loop,
// P reshaped via per-wave LDS.  Output fp32 [b, L, C].
// ---------------------------------------------------------------------------
__global__ void __launch_bounds__(256)
flash_attn_kernel(const bf16_t* __restrict__ qb,
                  const bf16_t* __restrict__ kb,
                  const bf16_t* __restrict__ vb,
                  float* __restrict__ ao)
{
    constexpr int BQ = 128, BK = 64;
    __shared__ __align__(32) bf16_t Ks[BK * HD_DIM];      // [key][c]   16KB
    __shared__ __align__(32) bf16_t Vs[HD_DIM * BK];      // [c][key]   16KB
    __shared__ __align__(32) bf16_t Ps[8 * 16 * BK];      // per-wave P 16KB

    const int tid  = threadIdx.x;
    const int lane = tid & 31;
    const int wave = tid >> 5;
    const int ln = lane & 15, hi = lane >> 4;
    const int qbase = blockIdx.x * BQ;
    const int head  = blockIdx.y;
    const int batch = blockIdx.z;
    const size_t bh = (size_t)batch * NH_DIM + head;

    // Q fragments for this wave's 16 rows: 4 K-chunks of 32 (whole hd=128)
    v16bf aq[4];
#pragma unroll
    for (int kk = 0; kk < 4; ++kk)
        aq[kk] = *(const v16bf*)(qb + ((bh * L_DIM + qbase + wave * 16 + ln) * HD_DIM)
                                     + kk * 32 + hi * 16);

    v8f oacc[8];
#pragma unroll
    for (int i = 0; i < 8; ++i) oacc[i] = v8f_zero();
    float mrun[8], lsum[8];
#pragma unroll
    for (int r = 0; r < 8; ++r) { mrun[r] = -3.0e38f; lsum[r] = 0.f; }

    bf16_t* myP = Ps + wave * 16 * BK;

    for (int kt = 0; kt < L_DIM / BK; ++kt) {
        const int keybase = kt * BK;
        __syncthreads();
#if HAVE_ASYNC_LDS
        {   // K tile: contiguous rows -> flat 16KB async copy
            const bf16_t* src = kb + (bh * L_DIM + keybase) * HD_DIM;
#pragma unroll
            for (int i = 0; i < 4; ++i) {
                int q = tid + i * 256;             // 1024 chunks of 8 bf16
                __builtin_amdgcn_global_load_async_to_lds_b128(
                    (as1_v4i*)(src + q * 8), (as3_v4i*)(Ks + q * 8), 0, 0);
            }
            // V tile: [c][key], strided rows of 128B
#pragma unroll
            for (int i = 0; i < 4; ++i) {
                int q = tid + i * 256;             // 1024 chunks
                int c = q >> 3, w = q & 7;
                __builtin_amdgcn_global_load_async_to_lds_b128(
                    (as1_v4i*)(vb + (bh * HD_DIM + c) * L_DIM + keybase + w * 8),
                    (as3_v4i*)(Vs + c * BK + w * 8), 0, 0);
            }
            __builtin_amdgcn_s_wait_asynccnt(0);
        }
#else
        {   // K tile: contiguous rows -> flat 16KB copy
            const uint4* src = (const uint4*)(kb + (bh * L_DIM + keybase) * HD_DIM);
            uint4* dst = (uint4*)Ks;
            for (int i = tid; i < BK * HD_DIM * 2 / 16; i += 256) dst[i] = src[i];
        }
        {   // V tile: [c][key], strided rows of 128B
            uint4* dst = (uint4*)Vs;
            for (int i = tid; i < HD_DIM * BK * 2 / 16; i += 256) {
                int c = i >> 3, w = i & 7;
                dst[i] = ((const uint4*)(vb + (bh * HD_DIM + c) * L_DIM + keybase))[w];
            }
        }
#endif
        __syncthreads();

        // S = Q * K^T  (16 x 64, already scaled via Q)
        v8f s[4];
#pragma unroll
        for (int nt = 0; nt < 4; ++nt) {
            v8f acc = v8f_zero();
#pragma unroll
            for (int kk = 0; kk < 4; ++kk) {
                v16bf bk = *(const v16bf*)(Ks + (nt * 16 + ln) * HD_DIM + kk * 32 + hi * 16);
                acc = wmma_bf16(aq[kk], bk, acc);
            }
            s[nt] = acc;
        }

        // Online softmax.  C-layout: VGPR r <-> row r + 8*hi, lane&15 <-> col.
#pragma unroll
        for (int r = 0; r < 8; ++r) {
            float lmax = fmaxf(fmaxf(s[0][r], s[1][r]), fmaxf(s[2][r], s[3][r]));
#pragma unroll
            for (int off = 1; off < 16; off <<= 1)
                lmax = fmaxf(lmax, __shfl_xor(lmax, off, 32));
            float mnew  = fmaxf(mrun[r], lmax);
            float scale = __expf(mrun[r] - mnew);
            mrun[r] = mnew;
            float rsum = 0.f;
#pragma unroll
            for (int nt = 0; nt < 4; ++nt) {
                float p = __expf(s[nt][r] - mnew);
                s[nt][r] = p;
                rsum += p;
            }
#pragma unroll
            for (int off = 1; off < 16; off <<= 1)
                rsum += __shfl_xor(rsum, off, 32);
            lsum[r] = lsum[r] * scale + rsum;
#pragma unroll
            for (int ot = 0; ot < 8; ++ot) oacc[ot][r] *= scale;
        }

        // P -> LDS in A-operand layout [row][key] (per-wave region, no barrier)
#pragma unroll
        for (int nt = 0; nt < 4; ++nt)
#pragma unroll
            for (int r = 0; r < 8; ++r)
                myP[(r + 8 * hi) * BK + nt * 16 + ln] = (bf16_t)s[nt][r];

        // O += P * V^T   (16 x 128, K = 64 keys in 2 chunks)
#pragma unroll
        for (int ot = 0; ot < 8; ++ot) {
#pragma unroll
            for (int kk = 0; kk < 2; ++kk) {
                v16bf ap = *(const v16bf*)(myP + ln * BK + kk * 32 + hi * 16);
                v16bf bv = *(const v16bf*)(Vs + (ot * 16 + ln) * BK + kk * 32 + hi * 16);
                oacc[ot] = wmma_bf16(ap, bv, oacc[ot]);
            }
        }
    }

    // Normalize and store to attn-out [b, L, C] fp32.
#pragma unroll
    for (int r = 0; r < 8; ++r) {
        float inv = 1.0f / lsum[r];
        int row = qbase + wave * 16 + r + 8 * hi;
#pragma unroll
        for (int ot = 0; ot < 8; ++ot) {
            int col = head * HD_DIM + ot * 16 + ln;
            ao[((size_t)batch * L_DIM + row) * C_DIM + col] = oacc[ot][r] * inv;
        }
    }
}

// ---------------------------------------------------------------------------
// Kernel 4: proj GEMM + bias + residual -> d_out fp32 [b, C, L].
// ---------------------------------------------------------------------------
__global__ void __launch_bounds__(256)
proj_gemm_kernel(const float* __restrict__ w_proj,
                 const float* __restrict__ ao,
                 const float* __restrict__ b_proj,
                 const float* __restrict__ x,
                 float* __restrict__ out)
{
    __shared__ __align__(32) bf16_t As[128 * 32];
    __shared__ __align__(32) bf16_t Bs[128 * 32];
    const int n0 = blockIdx.x * 128;
    const int m0 = blockIdx.y * 128;
    const int batch = blockIdx.z;

    v8f acc[2][4];
    gemm_tile(w_proj, ao + (size_t)batch * L_DIM * C_DIM, C_DIM, m0, n0, acc, As, Bs);

    const int lane = threadIdx.x & 31;
    const int wave = threadIdx.x >> 5;
    const int wm = wave >> 1, wn = wave & 1;
    const int ln = lane & 15, hi = lane >> 4;

#pragma unroll
    for (int sm = 0; sm < 2; ++sm)
#pragma unroll
        for (int sn = 0; sn < 4; ++sn)
#pragma unroll
            for (int r = 0; r < 8; ++r) {
                int o = m0 + wm * 32 + sm * 16 + r + 8 * hi;
                int l = n0 + wn * 64 + sn * 16 + ln;
                size_t idx = ((size_t)batch * C_DIM + o) * L_DIM + l;
                out[idx] = acc[sm][sn][r] + b_proj[o] + x[idx];
            }
}

// ---------------------------------------------------------------------------
extern "C" void kernel_launch(void* const* d_in, const int* in_sizes, int n_in,
                              void* d_out, int out_size, void* d_ws, size_t ws_size,
                              hipStream_t stream)
{
    const float* x      = (const float*)d_in[0];
    const float* gn_w   = (const float*)d_in[1];
    const float* gn_b   = (const float*)d_in[2];
    const float* w_qkv  = (const float*)d_in[3];
    const float* b_qkv  = (const float*)d_in[4];
    const float* w_proj = (const float*)d_in[5];
    const float* b_proj = (const float*)d_in[6];
    float* out = (float*)d_out;

    // Workspace layout (48 MB total):
    char* ws = (char*)d_ws;
    bf16_t* xn = (bf16_t*)(ws);                          //  8 MB  [b,L,C] bf16
    bf16_t* qb = (bf16_t*)(ws + ( 8u << 20));            //  8 MB  [b,nh,L,hd]
    bf16_t* kb = (bf16_t*)(ws + (16u << 20));            //  8 MB  [b,nh,L,hd]
    bf16_t* vb = (bf16_t*)(ws + (24u << 20));            //  8 MB  [b,nh,hd,L]
    float*  ao = (float*) (ws + (32u << 20));            // 16 MB  [b,L,C] f32

    groupnorm_kernel<<<dim3(32, B_DIM), 256, 0, stream>>>(x, gn_w, gn_b, xn);

    qkv_gemm_kernel<<<dim3(L_DIM / 128, (3 * C_DIM) / 128, B_DIM), 256, 0, stream>>>(
        w_qkv, xn, b_qkv, qb, kb, vb);

    flash_attn_kernel<<<dim3(L_DIM / 128, NH_DIM, B_DIM), 256, 0, stream>>>(
        qb, kb, vb, ao);

    proj_gemm_kernel<<<dim3(L_DIM / 128, C_DIM / 128, B_DIM), 256, 0, stream>>>(
        w_proj, ao, b_proj, x, out);
}
